// CrossAttentionBlock_57002805953294
// MI455X (gfx1250) — matrix operand order repaired and checked by reading, not compiled, not run on hardware
//
#include <hip/hip_runtime.h>
#include <hip/hip_bf16.h>

// CDNA5 / gfx1250 implementation of the video<->audio windowed cross-attention
// block. All heavy math goes through v_wmma_f32_16x16x32_bf16 (wave32 WMMA).
// Intermediates are bf16 to halve HBM traffic; attention is fused flash-style
// so scores never hit HBM. Workspace use: ~109 MB.

typedef __bf16 bf16_t;
typedef __attribute__((ext_vector_type(16))) __bf16 v16bf;
typedef __attribute__((ext_vector_type(8)))  __bf16 v8bf_t;   // one 16-byte load
typedef __attribute__((ext_vector_type(8)))  float  v8f;

union FragU { v16bf f; v8bf_t h[2]; };

__device__ __forceinline__ int lane_id() { return (int)(threadIdx.x & 31u); }

__device__ __forceinline__ v8f zero8() {
  v8f z = {0.f,0.f,0.f,0.f,0.f,0.f,0.f,0.f};
  return z;
}

// A fragment: 16x32 bf16, row-major source, row m = lane&15.
// element e <-> K = (lane>>4)*16 + e  => one contiguous 32B run (two 16B loads).
__device__ __forceinline__ v16bf load_a_frag(const bf16_t* base, int lda) {
  int l = lane_id(); int m = l & 15, h = l >> 4;
  const bf16_t* row = base + (size_t)m * lda + h * 16;
  FragU u;
  u.h[0] = *(const v8bf_t*)(row);
  u.h[1] = *(const v8bf_t*)(row + 8);
  return u.f;
}

// B fragment: 32x16 bf16 from Bt[N][K] row-major, column n = lane&15.
// element e <-> K = (lane>>4)*16 + e => one contiguous 32B run (two 16B loads).
__device__ __forceinline__ v16bf load_b_frag(const bf16_t* base, int ldb) {
  int l = lane_id(); int n = l & 15, h = l >> 4;
  const bf16_t* row = base + (size_t)n * ldb + h * 16;
  FragU u;
  u.h[0] = *(const v8bf_t*)(row);
  u.h[1] = *(const v8bf_t*)(row + 8);
  return u.f;
}

__device__ __forceinline__ v8f wmma_bf16(v16bf a, v16bf b, v8f c) {
  return __builtin_amdgcn_wmma_f32_16x16x32_bf16(
      /*neg_a=*/false, a, /*neg_b=*/false, b,
      /*c_mod=*/(short)0, c, /*reuse_a=*/false, /*reuse_b=*/false);
}

// ---------------------------------------------------------------------------
// GroupNorm statistics: one block per group, writes {mu, rsqrt(var+eps)}.
// ---------------------------------------------------------------------------
__global__ __launch_bounds__(256) void gn_stats_video(const float* __restrict__ v,
                                                      float* __restrict__ stats) {
  int g = blockIdx.x;
  float s = 0.f, ss = 0.f;
  for (int i = threadIdx.x; i < 16 * 16384; i += 256) {
    int ch = g * 16 + (i >> 14);
    int t  = i & 16383;
    float x = v[(size_t)(t >> 10) * 524288 + (size_t)ch * 1024 + (t & 1023)];
    s += x; ss += x * x;
  }
#pragma unroll
  for (int off = 16; off; off >>= 1) { s += __shfl_down(s, off, 32); ss += __shfl_down(ss, off, 32); }
  __shared__ float b1[8], b2[8];
  int wid = threadIdx.x >> 5;
  if ((threadIdx.x & 31) == 0) { b1[wid] = s; b2[wid] = ss; }
  __syncthreads();
  if (threadIdx.x == 0) {
    float S = 0.f, SS = 0.f;
    for (int i = 0; i < 8; i++) { S += b1[i]; SS += b2[i]; }
    float inv = 1.f / 262144.f;
    float mu = S * inv, var = SS * inv - mu * mu;
    stats[2 * g] = mu; stats[2 * g + 1] = rsqrtf(var + 1e-5f);
  }
}

__global__ __launch_bounds__(256) void gn_stats_audio(const float* __restrict__ a,
                                                      float* __restrict__ stats) {
  int g = blockIdx.x;
  const float* base = a + (size_t)g * 65536;
  float s = 0.f, ss = 0.f;
  for (int i = threadIdx.x; i < 65536; i += 256) { float x = base[i]; s += x; ss += x * x; }
#pragma unroll
  for (int off = 16; off; off >>= 1) { s += __shfl_down(s, off, 32); ss += __shfl_down(ss, off, 32); }
  __shared__ float b1[8], b2[8];
  int wid = threadIdx.x >> 5;
  if ((threadIdx.x & 31) == 0) { b1[wid] = s; b2[wid] = ss; }
  __syncthreads();
  if (threadIdx.x == 0) {
    float S = 0.f, SS = 0.f;
    for (int i = 0; i < 8; i++) { S += b1[i]; SS += b2[i]; }
    float inv = 1.f / 65536.f;
    float mu = S * inv, var = SS * inv - mu * mu;
    stats[2 * g] = mu; stats[2 * g + 1] = rsqrtf(var + 1e-5f);
  }
}

// ---------------------------------------------------------------------------
// Normalize + transpose to token-major bf16  X_t[t][512]
// ---------------------------------------------------------------------------
__global__ __launch_bounds__(256) void norm_video(const float* __restrict__ v,
                                                  const float* __restrict__ stats,
                                                  const float* __restrict__ gamma,
                                                  const float* __restrict__ beta,
                                                  bf16_t* __restrict__ xt) {
  int t = blockIdx.x; int f = t >> 10, hw = t & 1023;
  for (int c = threadIdx.x; c < 512; c += 256) {
    int g = c >> 4;
    float x = v[(size_t)f * 524288 + (size_t)c * 1024 + hw];
    float y = (x - stats[2 * g]) * stats[2 * g + 1] * gamma[c] + beta[c];
    xt[(size_t)t * 512 + c] = (bf16_t)y;
  }
}

__global__ __launch_bounds__(256) void norm_audio(const float* __restrict__ a,
                                                  const float* __restrict__ stats,
                                                  const float* __restrict__ gamma,
                                                  const float* __restrict__ beta,
                                                  bf16_t* __restrict__ xt) {
  int t = blockIdx.x;
  for (int c = threadIdx.x; c < 512; c += 256) {
    int g = c >> 4;
    float x = a[(size_t)c * 4096 + t];
    float y = (x - stats[2 * g]) * stats[2 * g + 1] * gamma[c] + beta[c];
    xt[(size_t)t * 512 + c] = (bf16_t)y;
  }
}

__global__ __launch_bounds__(256) void cvt_bf16(const float* __restrict__ in,
                                                bf16_t* __restrict__ out, int n) {
  int i = blockIdx.x * 256 + threadIdx.x;
  if (i < n) out[i] = (bf16_t)in[i];
}

// ---------------------------------------------------------------------------
// Generic WMMA GEMM:  C[M][N] = A[M][K] * Bt[N][K]  (+ epilogue)
// block = 256 threads = 8 waves (2 x 4), wave tile 32x64, block tile 64x256.
// MODE 0: bf16 out, bias over N          (QKV -> token-major QK_t)
// MODE 1: bf16 out, bias over M          (V   -> channel-major Vc)
// MODE 2: f32 out + resid, bias over M   (audio projection)
// MODE 3: f32 video scatter + resid      (video projection, n = token)
// ---------------------------------------------------------------------------
template<int MODE>
__global__ __launch_bounds__(256) void gemm_kernel(
    const bf16_t* __restrict__ A, int lda,
    const bf16_t* __restrict__ Bt, int ldb,
    void* __restrict__ Cout, long ldc,
    const float* __restrict__ bias,
    const float* __restrict__ resid,
    int K)
{
  int w  = threadIdx.x >> 5;
  int wm = w >> 2, wn = w & 3;
  int m0 = blockIdx.y * 64  + wm * 32;
  int n0 = blockIdx.x * 256 + wn * 64;

  v8f acc[2][4];
#pragma unroll
  for (int i = 0; i < 2; i++)
#pragma unroll
    for (int j = 0; j < 4; j++) acc[i][j] = zero8();

  for (int k0 = 0; k0 < K; k0 += 32) {
    v16bf af[2], bfr[4];
#pragma unroll
    for (int i = 0; i < 2; i++)
      af[i] = load_a_frag(A + (size_t)(m0 + 16 * i) * lda + k0, lda);
#pragma unroll
    for (int j = 0; j < 4; j++)
      bfr[j] = load_b_frag(Bt + (size_t)(n0 + 16 * j) * ldb + k0, ldb);
#pragma unroll
    for (int i = 0; i < 2; i++)
#pragma unroll
      for (int j = 0; j < 4; j++)
        acc[i][j] = wmma_bf16(af[i], bfr[j], acc[i][j]);
  }

  int l = lane_id(); int ln = l & 15, h = l >> 4;
#pragma unroll
  for (int i = 0; i < 2; i++) {
#pragma unroll
    for (int j = 0; j < 4; j++) {
#pragma unroll
      for (int r = 0; r < 8; r++) {
        int m = m0 + 16 * i + r + h * 8;
        int n = n0 + 16 * j + ln;
        float val = acc[i][j][r];
        if (MODE == 0) {
          ((bf16_t*)Cout)[(size_t)m * ldc + n] = (bf16_t)(val + bias[n]);
        } else if (MODE == 1) {
          ((bf16_t*)Cout)[(size_t)m * ldc + n] = (bf16_t)(val + bias[m]);
        } else if (MODE == 2) {
          size_t idx = (size_t)m * ldc + n;
          ((float*)Cout)[idx] = resid[idx] + val + bias[m];
        } else {  // MODE 3: n is a video token -> (f, hw) scatter
          int f = n >> 10, hw = n & 1023;
          size_t idx = (size_t)f * 524288 + (size_t)m * 1024 + hw;
          ((float*)Cout)[idx] = resid[idx] + val + bias[m];
        }
      }
    }
  }
}

// ---------------------------------------------------------------------------
// Fused windowed attention (flash-style online softmax).
// SIDE=0: video queries (S=512 audio window), SIDE=1: audio queries (S=2048).
// One wave handles 16 query tokens x full head; block = 8 waves = 128 rows.
// ---------------------------------------------------------------------------
template<int SIDE>
__global__ __launch_bounds__(256) void attn_kernel(
    const bf16_t* __restrict__ qkt,   // [20480][1024]  (q: 0..511, k: 512..1023)
    const bf16_t* __restrict__ vc,    // [512][20480]   channel-major V
    bf16_t* __restrict__ outt)        // [20480][512]   token-major O
{
  __shared__ bf16_t plds[8][16 * 32];   // per-wave P tile (C-layout -> A-layout)
  int wid  = threadIdx.x >> 5;
  int lane = lane_id(); int ln = lane & 15, h = lane >> 4;
  int frame = blockIdx.y, head = blockIdx.z;

  const int S        = SIDE ? 2048  : 512;
  const int win_off  = SIDE ? 0     : 16384;
  const int win_mask = SIDE ? 16383 : 4095;
  const int win_mul  = SIDE ? 1024  : 256;
  int q_tok0 = (SIDE ? 16384 + frame * 256 : frame * 1024) + blockIdx.x * 128 + wid * 16;
  int qcol = head * 64;
  int kcol = 512 + head * 64;

  v16bf aq[2];
#pragma unroll
  for (int i = 0; i < 2; i++)
    aq[i] = load_a_frag(qkt + (size_t)q_tok0 * 1024 + qcol + 32 * i, 1024);

  v8f o_acc[4];
#pragma unroll
  for (int j = 0; j < 4; j++) o_acc[j] = zero8();
  float mrow[8], lrow[8];
#pragma unroll
  for (int r = 0; r < 8; r++) { mrow[r] = -3.0e38f; lrow[r] = 0.f; }

  for (int cs = 0; cs < S; cs += 32) {
    // ---- S = Q * K^T  (16 x 32), k over ch=64 ----
    v8f s_acc[2];
#pragma unroll
    for (int j = 0; j < 2; j++) s_acc[j] = zero8();
#pragma unroll
    for (int j = 0; j < 2; j++) {
      int stok = (frame * win_mul + cs + 16 * j + ln) & win_mask;
      const bf16_t* krow = qkt + (size_t)(win_off + stok) * 1024 + kcol;
#pragma unroll
      for (int kk = 0; kk < 2; kk++) {
        const bf16_t* p = krow + 32 * kk + h * 16;
        FragU u;
        u.h[0] = *(const v8bf_t*)(p);
        u.h[1] = *(const v8bf_t*)(p + 8);
        s_acc[j] = wmma_bf16(aq[kk], u.f, s_acc[j]);
      }
    }
    // ---- online softmax (row r+h*8 lives in one 16-lane half) ----
#pragma unroll
    for (int r = 0; r < 8; r++) {
      float a0 = s_acc[0][r] * 0.125f;          // scale^2 = ch^-0.5 = 1/8
      float a1 = s_acc[1][r] * 0.125f;
      float mx = fmaxf(a0, a1);
#pragma unroll
      for (int off = 1; off < 16; off <<= 1) mx = fmaxf(mx, __shfl_xor(mx, off, 32));
      float mnew  = fmaxf(mrow[r], mx);
      float alpha = __expf(mrow[r] - mnew);
      float p0 = __expf(a0 - mnew), p1 = __expf(a1 - mnew);
      float sum = p0 + p1;
#pragma unroll
      for (int off = 1; off < 16; off <<= 1) sum += __shfl_xor(sum, off, 32);
      lrow[r] = lrow[r] * alpha + sum;
      mrow[r] = mnew;
#pragma unroll
      for (int j = 0; j < 4; j++) o_acc[j][r] *= alpha;
      int prow = r + h * 8;
      plds[wid][prow * 32 + ln]      = (bf16_t)p0;
      plds[wid][prow * 32 + 16 + ln] = (bf16_t)p1;
    }
    asm volatile("s_wait_dscnt 0x0" ::: "memory");  // per-wave LDS RAW fence
    v16bf pf = load_a_frag(&plds[wid][0], 32);
    // ---- O += P * V  (16x32 x 32x64); V window is contiguous, 16-aligned ----
    int vt0 = (frame * win_mul + cs + h * 16) & win_mask;
#pragma unroll
    for (int j = 0; j < 4; j++) {
      int row_c = head * 64 + 16 * j + ln;
      const bf16_t* p = vc + (size_t)row_c * 20480 + win_off + vt0;
      FragU u;
      u.h[0] = *(const v8bf_t*)(p);
      u.h[1] = *(const v8bf_t*)(p + 8);
      o_acc[j] = wmma_bf16(pf, u.f, o_acc[j]);
    }
  }

  // ---- normalize + store token-major bf16 ----
#pragma unroll
  for (int j = 0; j < 4; j++) {
#pragma unroll
    for (int r = 0; r < 8; r++) {
      float val = o_acc[j][r] / lrow[r];
      int t = q_tok0 + r + h * 8;
      int c = head * 64 + 16 * j + ln;
      outt[(size_t)t * 512 + c] = (bf16_t)val;
    }
  }
}

// ---------------------------------------------------------------------------
extern "C" void kernel_launch(void* const* d_in, const int* in_sizes, int n_in,
                              void* d_out, int out_size, void* d_ws, size_t ws_size,
                              hipStream_t stream) {
  (void)in_sizes; (void)n_in; (void)out_size; (void)ws_size;
  const float* video   = (const float*)d_in[0];
  const float* audio   = (const float*)d_in[1];
  const float* v_gam   = (const float*)d_in[2];
  const float* v_bet   = (const float*)d_in[3];
  const float* a_gam   = (const float*)d_in[4];
  const float* a_bet   = (const float*)d_in[5];
  const float* v_qkv_w = (const float*)d_in[6];
  const float* v_qkv_b = (const float*)d_in[7];
  const float* a_qkv_w = (const float*)d_in[8];
  const float* a_qkv_b = (const float*)d_in[9];
  const float* vproj_w = (const float*)d_in[10];
  const float* vproj_b = (const float*)d_in[11];
  const float* aproj_w = (const float*)d_in[12];
  const float* aproj_b = (const float*)d_in[13];

  // workspace carve-up (256B aligned); total ~109 MB
  char* ws = (char*)d_ws;
  size_t off = 0;
  auto carve = [&](size_t bytes) { char* p = ws + off; off += (bytes + 255) & ~(size_t)255; return p; };
  float*  stats_v = (float*) carve(64 * sizeof(float));
  float*  stats_a = (float*) carve(64 * sizeof(float));
  bf16_t* Wv  = (bf16_t*)carve((size_t)1536 * 512 * 2);   // v_qkv_w bf16
  bf16_t* Wa  = (bf16_t*)carve((size_t)1536 * 512 * 2);   // a_qkv_w bf16
  bf16_t* Pv  = (bf16_t*)carve((size_t)512 * 512 * 2);    // video_proj_w bf16
  bf16_t* Pa  = (bf16_t*)carve((size_t)512 * 512 * 2);    // audio_proj_w bf16
  bf16_t* Xv  = (bf16_t*)carve((size_t)16384 * 512 * 2);  // normed video tokens [t][c]
  bf16_t* Xa  = (bf16_t*)carve((size_t)4096  * 512 * 2);  // normed audio tokens [t][c]
  bf16_t* QK  = (bf16_t*)carve((size_t)20480 * 1024 * 2); // [t][q:512|k:512]
  bf16_t* Vc  = (bf16_t*)carve((size_t)512 * 20480 * 2);  // [o][t]
  bf16_t* Ot  = (bf16_t*)carve((size_t)20480 * 512 * 2);  // attention out [t][c]

  // 1) group-norm stats
  gn_stats_video<<<32, 256, 0, stream>>>(video, stats_v);
  gn_stats_audio<<<32, 256, 0, stream>>>(audio, stats_a);

  // 2) normalize + token-major bf16
  norm_video<<<16384, 256, 0, stream>>>(video, stats_v, v_gam, v_bet, Xv);
  norm_audio<<<4096,  256, 0, stream>>>(audio, stats_a, a_gam, a_bet, Xa);

  // 3) weights -> bf16
  cvt_bf16<<<(1536 * 512 + 255) / 256, 256, 0, stream>>>(v_qkv_w, Wv, 1536 * 512);
  cvt_bf16<<<(1536 * 512 + 255) / 256, 256, 0, stream>>>(a_qkv_w, Wa, 1536 * 512);
  cvt_bf16<<<(512 * 512 + 255) / 256, 256, 0, stream>>>(vproj_w, Pv, 512 * 512);
  cvt_bf16<<<(512 * 512 + 255) / 256, 256, 0, stream>>>(aproj_w, Pa, 512 * 512);

  // 4) QK GEMMs: QK_t[t][o] = X_t[t][c] * Wqk[o][c] + b[o]
  gemm_kernel<0><<<dim3(4, 256), 256, 0, stream>>>(Xv, 512, Wv, 512, QK, 1024, v_qkv_b, nullptr, 512);
  gemm_kernel<0><<<dim3(4, 64),  256, 0, stream>>>(Xa, 512, Wa, 512, QK + (size_t)16384 * 1024, 1024, a_qkv_b, nullptr, 512);

  // 5) V GEMMs (channel-major): Vc[o][t] = Wv_v[o][c] * X_t[t][c] + b[o]
  gemm_kernel<1><<<dim3(64, 8), 256, 0, stream>>>(Wv + (size_t)1024 * 512, 512, Xv, 512, Vc, 20480, v_qkv_b + 1024, nullptr, 512);
  gemm_kernel<1><<<dim3(16, 8), 256, 0, stream>>>(Wa + (size_t)1024 * 512, 512, Xa, 512, Vc + 16384, 20480, a_qkv_b + 1024, nullptr, 512);

  // 6) fused windowed attention (flash softmax)
  attn_kernel<0><<<dim3(8, 16, 8), 256, 0, stream>>>(QK, Vc, Ot);  // video queries
  attn_kernel<1><<<dim3(2, 16, 8), 256, 0, stream>>>(QK, Vc, Ot);  // audio queries

  // 7) output projections + residual (f32)
  gemm_kernel<3><<<dim3(64, 8), 256, 0, stream>>>(Pv, 512, Ot, 512, d_out, 0, vproj_b, video, 512);
  gemm_kernel<2><<<dim3(16, 8), 256, 0, stream>>>(Pa, 512, Ot + (size_t)16384 * 512, 512,
                                                  (float*)d_out + 8388608, 4096, aproj_b, audio, 512);
}
